// KANLinear_17738214932721
// MI455X (gfx1250) — compile-verified, matrix operand order checked by reading, not run on hardware
//
#include <hip/hip_runtime.h>
#include <hip/hip_bf16.h>
#include <math.h>

// ---------------------------------------------------------------------------
// KAN linear layer, fused, for MI455X (gfx1250, wave32, WMMA + TDM).
//
// One GEMM with expanded K = 512*(1+8+10) = 9728:
//   - prep kernel packs weights(+scalers) into bf16 Wmega[OUT_F][9728] in d_ws
//   - fused GEMM: W tiles DMA'd into LDS by the Tensor Data Mover
//     (tensor_load_to_lds, TENSORcnt), A tiles (feature expansion of x)
//     generated in-kernel (divide-free), double-buffered LDS pipeline,
//     accumulation with v_wmma_f32_16x16x32_bf16.
// ---------------------------------------------------------------------------

#define IN_F        512
#define OUT_F       512
#define NBASIS      10
#define KOFF_SPLINE 512          // [0,512): base (silu)
#define KOFF_LOGI   4608         // [512,4608): spline, [4608,9728): logistic
#define KTOT        9728

#define BM   128                 // rows (batch) per block
#define BN   256                 // cols (out features) per block
#define KB   64                  // expanded-K per LDS chunk (2 WMMA k-steps)
#define LDK  72                  // padded LDS row stride (bf16); 144B = 9*16B
#define A_ELEMS (BM * LDK)       // 9216 halfs per A buffer
#define W_ELEMS (BN * LDK)       // 18432 halfs per W buffer

typedef __bf16 v16bf __attribute__((ext_vector_type(16)));
typedef __bf16 v8bf  __attribute__((ext_vector_type(8)));
typedef float  v8f   __attribute__((ext_vector_type(8)));
typedef unsigned int v4u __attribute__((ext_vector_type(4)));
typedef unsigned int v8u __attribute__((ext_vector_type(8)));

__device__ __forceinline__ unsigned short f32_bf16(float f) {
    union { float f; unsigned u; } v; v.f = f;
    unsigned r = v.u + 0x7FFFu + ((v.u >> 16) & 1u);   // round-to-nearest-even
    return (unsigned short)(r >> 16);
}

// Load one 16x32 bf16 WMMA operand fragment from an LDS row laid out [M|N][K].
// lanes 0-15: khalf=0 (K 0-7,16-23), lanes 16-31: khalf=8 (K 8-15,24-31),
// per CDNA5 ISA 7.12.2 16-bit operand layout.
__device__ __forceinline__ v16bf load_frag(const unsigned short* p) {
    v8bf lo = *reinterpret_cast<const v8bf*>(p);
    v8bf hi = *reinterpret_cast<const v8bf*>(p + 16);
    return __builtin_shufflevector(lo, hi, 0,1,2,3,4,5,6,7,8,9,10,11,12,13,14,15);
}

// ---------------------------------------------------------------------------
// TDM: DMA a BN x KB bf16 tile (row stride KTOT elems) from global into LDS,
// inserting 16B padding after each 128B row -> LDS layout [BN][LDK] bf16.
// 2D tensor: descriptor groups 0+1 only (VADDR2/3 = NULL form).
// ---------------------------------------------------------------------------
__device__ __forceinline__ void tdm_load_w_tile(const unsigned short* gsrc,
                                                unsigned lds_off_bytes) {
    unsigned long long ga = (unsigned long long)(size_t)gsrc;
    v4u g0;
    g0[0] = 1u;                                   // count=1 (valid), user mode
    g0[1] = lds_off_bytes;                        // lds_addr
    g0[2] = (unsigned)(ga & 0xFFFFFFFFu);         // global_addr[31:0]
    g0[3] = (unsigned)((ga >> 32) & 0x01FFFFFFu)  // global_addr[56:32]
          | (2u << 30);                           // type=2 ("image")
    v8u g1;
    g1[0] = (1u << 16)        // data_size = 1 -> 2 bytes
          | (1u << 20)        // pad_enable
          | (4u << 22)        // pad_interval = 4 -> every 32 DWORDs (128B row)
          | (3u << 25);       // pad_amount   = 3 -> 4 DWORDs (16B)
    g1[1] = ((unsigned)KTOT & 0xFFFFu) << 16;     // tensor_dim0[15:0]
    g1[2] = ((unsigned)KTOT >> 16)                // tensor_dim0[31:16]
          | ((unsigned)OUT_F << 16);              // tensor_dim1[15:0]
    g1[3] = ((unsigned)KB << 16);                 // tensor_dim1 hi=0 | tile_dim0=64
    g1[4] = (unsigned)BN;                         // tile_dim1=256, tile_dim2=0
    g1[5] = (unsigned)KTOT;                       // tensor_dim0_stride[31:0]
    g1[6] = 0u;                                   // stride hi / dim1_stride lo
    g1[7] = 0u;
    asm volatile("tensor_load_to_lds %0, %1" : : "s"(g0), "s"(g1) : "memory");
}

// ---------------------------------------------------------------------------
// Generate one BM x KB bf16 A-tile (feature expansion of x) into LDS.
// All divides removed: knot reciprocals constant-folded, v_rcp_f32 for the
// sigmoid denominators (feeding a bf16 GEMM, approx rcp is plenty).
// ---------------------------------------------------------------------------
__device__ __forceinline__ void gen_A_tile(unsigned short* ldsA,
                                           const float* __restrict__ x,
                                           const float* __restrict__ aP,
                                           const float* __restrict__ bP,
                                           int rowBase, int kB, int tid) {
    if (kB < KOFF_SPLINE) {
        // base region: A[b][k] = silu(x[b][k])
        #pragma unroll 8
        for (int t = 0; t < (BM * KB) / 256; ++t) {
            int idx = tid + t * 256;
            int r = idx >> 6, c = idx & 63;
            float xv = x[(size_t)(rowBase + r) * IN_F + kB + c];
            float s  = xv * __builtin_amdgcn_rcpf(1.0f + __expf(-xv));
            ldsA[r * LDK + c] = f32_bf16(s);
        }
    } else if (kB < KOFF_LOGI) {
        // spline region: 8 features/chunk; one Cox-de-Boor recurrence -> 8 bases
        int iBase = (kB - KOFF_SPLINE) >> 3;
        #pragma unroll 2
        for (int t = 0; t < (BM * 8) / 256; ++t) {
            int pidx = tid + t * 256;
            int r = pidx >> 3, fi = pidx & 7;
            float xv = x[(size_t)(rowBase + r) * IN_F + iBase + fi];
            float kn[12];
            #pragma unroll
            for (int j = 0; j < 12; ++j) kn[j] = -1.0f + 0.4f * (float)(j - 3);
            float bas[11];
            #pragma unroll
            for (int j = 0; j < 11; ++j)
                bas[j] = (xv >= kn[j] && xv < kn[j + 1]) ? 1.0f : 0.0f;
            #pragma unroll
            for (int k = 1; k <= 3; ++k) {
                #pragma unroll
                for (int j = 0; j + k < 11; ++j) {
                    // reciprocals of constant knot spans -> compile-time consts
                    float r0 = 1.0f / (kn[j + k]     - kn[j]);
                    float r1 = 1.0f / (kn[j + k + 1] - kn[j + 1]);
                    bas[j] = (xv - kn[j]) * r0 * bas[j]
                           + (kn[j + k + 1] - xv) * r1 * bas[j + 1];
                }
            }
            #pragma unroll
            for (int g = 0; g < 8; ++g)
                ldsA[r * LDK + fi * 8 + g] = f32_bf16(bas[g]);
        }
    } else {
        // logistic region: A[b][k] = 2 / (1 + exp(-a*(x-b)))
        #pragma unroll 4
        for (int t = 0; t < (BM * KB) / 256; ++t) {
            int idx = tid + t * 256;
            int r = idx >> 6, c = idx & 63;
            int kg = kB + c - KOFF_LOGI;
            int i  = kg / 10, n = kg - i * 10;
            float xv = x[(size_t)(rowBase + r) * IN_F + i];
            float av = aP[i * NBASIS + n];
            float bv = bP[i * NBASIS + n];
            float ph = 2.0f * __builtin_amdgcn_rcpf(1.0f + __expf(-av * (xv - bv)));
            ldsA[r * LDK + c] = f32_bf16(ph);
        }
    }
}

// ---------------------------------------------------------------------------
// Kernel 1: fold weights + scalers into bf16 Wmega[OUT_F][KTOT] ([N][K] layout).
// ---------------------------------------------------------------------------
__global__ __launch_bounds__(256)
void kan_prep_weights(const float* __restrict__ base_w,
                      const float* __restrict__ spline_w,
                      const float* __restrict__ spline_s,
                      const float* __restrict__ logi_w,
                      const float* __restrict__ logi_s,
                      unsigned short* __restrict__ wmega) {
    int e = blockIdx.x * blockDim.x + threadIdx.x;
    if (e >= OUT_F * KTOT) return;
    int o = e / KTOT;
    int k = e - o * KTOT;
    float v;
    if (k < KOFF_SPLINE) {
        v = base_w[o * IN_F + k];
    } else if (k < KOFF_LOGI) {
        int j = k - KOFF_SPLINE;
        int i = j >> 3, g = j & 7;
        v = spline_w[(o * IN_F + i) * 8 + g] * spline_s[o * IN_F + i];
    } else {
        int j = k - KOFF_LOGI;                    // SCALE_LOGISTIC == 1.0
        v = logi_w[o * (IN_F * NBASIS) + j] * logi_s[o];
    }
    wmega[e] = f32_bf16(v);
}

// ---------------------------------------------------------------------------
// Kernel 2: fused feature-expansion + GEMM, double-buffered, TDM W staging.
// 256 threads (8 wave32) -> BM x BN = 128x256 output tile.
// Wave w: rows [(w&3)*32,+32), cols [(w>>2)*128,+128) = 2x8 WMMA tiles.
// ---------------------------------------------------------------------------
__global__ __launch_bounds__(256)
void kan_fused_gemm(const float* __restrict__ x,
                    const float* __restrict__ aP,
                    const float* __restrict__ bP,
                    const unsigned short* __restrict__ wmega,
                    float* __restrict__ out) {
    __shared__ __align__(64) unsigned short ldsA[2 * A_ELEMS];   // 36 KB
    __shared__ __align__(64) unsigned short ldsW[2 * W_ELEMS];   // 72 KB

    const int rowBase = blockIdx.x * BM;
    const int colBase = blockIdx.y * BN;
    const int tid  = threadIdx.x;
    const int wave = tid >> 5;
    const int lane = tid & 31;
    const int wm = wave & 3;            // 4 row groups of 32
    const int wn = wave >> 2;           // 2 col groups of 128
    const int lrow  = lane & 15;
    const int khalf = (lane >> 4) << 3; // 0 or 8 (ISA operand split)

    const unsigned ldsWbase = (unsigned)(size_t)(&ldsW[0]);

    v8f acc[2][8];
    #pragma unroll
    for (int mi = 0; mi < 2; ++mi)
        #pragma unroll
        for (int ni = 0; ni < 8; ++ni) {
            v8f z = {0.f,0.f,0.f,0.f,0.f,0.f,0.f,0.f};
            acc[mi][ni] = z;
        }

    const int NCHUNK = KTOT / KB;       // 152

    // ---- prologue: stage chunk 0 into buffer 0 ----
    if (wave == 0)
        tdm_load_w_tile(wmega + (size_t)colBase * KTOT, ldsWbase);
    gen_A_tile(&ldsA[0], x, aP, bP, rowBase, 0, tid);
    __builtin_amdgcn_s_wait_tensorcnt(0);
    __syncthreads();

    for (int ic = 0; ic < NCHUNK; ++ic) {
        const int cur = ic & 1;
        const int nxt = cur ^ 1;

        // ---- stage chunk ic+1 into the spare buffer (overlaps WMMA below) ----
        if (ic + 1 < NCHUNK) {
            if (wave == 0)
                tdm_load_w_tile(wmega + (size_t)colBase * KTOT + (size_t)(ic + 1) * KB,
                                ldsWbase + (unsigned)nxt * W_ELEMS * 2u);
            gen_A_tile(&ldsA[nxt * A_ELEMS], x, aP, bP, rowBase, (ic + 1) * KB, tid);
        }

        // ---- WMMA on current buffer: 2 k-steps x (2x8 tiles) = 32 wmma ----
        // Explicit 3-register rotation on the B fragment: b0 consumed, b1
        // staged, bn prefetched 2 iterations ahead. All three are live across
        // the WMMA pair by data dependence, so the allocator cannot collapse
        // them and LDS latency overlaps the XDL pipe.
        const unsigned short* bufA = &ldsA[cur * A_ELEMS];
        const unsigned short* bufW = &ldsW[cur * W_ELEMS];
        #pragma unroll
        for (int kk = 0; kk < KB; kk += 32) {
            v16bf aF[2];
            #pragma unroll
            for (int mi = 0; mi < 2; ++mi)
                aF[mi] = load_frag(bufA + (wm * 32 + mi * 16 + lrow) * LDK + kk + khalf);

            const unsigned short* wrow = bufW + (wn * 128 + lrow) * LDK + kk + khalf;
            v16bf b0 = load_frag(wrow);
            v16bf b1 = load_frag(wrow + 16 * LDK);
            #pragma unroll
            for (int ni = 0; ni < 8; ++ni) {
                v16bf bn;
                if (ni + 2 < 8)
                    bn = load_frag(wrow + (ni + 2) * 16 * LDK);
                acc[0][ni] = __builtin_amdgcn_wmma_f32_16x16x32_bf16(
                    false, aF[0], false, b0, (short)0, acc[0][ni], false, false);
                acc[1][ni] = __builtin_amdgcn_wmma_f32_16x16x32_bf16(
                    false, aF[1], false, b0, (short)0, acc[1][ni], false, false);
                b0 = b1;
                if (ni + 2 < 8) b1 = bn;
            }
        }

        __builtin_amdgcn_s_wait_tensorcnt(0);   // TDM for chunk ic+1 landed
        __syncthreads();                        // one barrier per chunk
    }

    // ---- store f32 accumulators (C layout: VGPR v -> M=v / v+8) ----
    #pragma unroll
    for (int mi = 0; mi < 2; ++mi) {
        #pragma unroll
        for (int ni = 0; ni < 8; ++ni) {
            int col   = colBase + wn * 128 + ni * 16 + lrow;
            int rbase = rowBase + wm * 32 + mi * 16 + ((lane >> 4) << 3);
            #pragma unroll
            for (int v = 0; v < 8; ++v)
                out[(size_t)(rbase + v) * OUT_F + col] = acc[mi][ni][v];
        }
    }
}

// ---------------------------------------------------------------------------
extern "C" void kernel_launch(void* const* d_in, const int* in_sizes, int n_in,
                              void* d_out, int out_size, void* d_ws, size_t ws_size,
                              hipStream_t stream) {
    const float* x   = (const float*)d_in[0];   // [B, IN_F]
    const float* bw  = (const float*)d_in[1];   // [OUT_F, IN_F]
    const float* sw  = (const float*)d_in[2];   // [OUT_F, IN_F, 8]
    const float* ss  = (const float*)d_in[3];   // [OUT_F, IN_F]
    const float* aP  = (const float*)d_in[4];   // [IN_F, 10]
    const float* bP  = (const float*)d_in[5];   // [IN_F, 10]
    const float* lw  = (const float*)d_in[6];   // [OUT_F, IN_F*10]
    const float* ls  = (const float*)d_in[7];   // [OUT_F]
    float* out = (float*)d_out;

    unsigned short* wmega = (unsigned short*)d_ws;   // bf16 [OUT_F][KTOT], ~9.96MB

    int wtotal = OUT_F * KTOT;
    kan_prep_weights<<<(wtotal + 255) / 256, 256, 0, stream>>>(bw, sw, ss, lw, ls, wmega);

    int Brows = in_sizes[0] / IN_F;              // 16384
    dim3 grid(Brows / BM, OUT_F / BN);           // (128, 2)
    kan_fused_gemm<<<grid, 256, 0, stream>>>(x, aP, bP, wmega, out);
}
